// TransformerBlock_21079699488997
// MI455X (gfx1250) — matrix operand order repaired and checked by reading, not compiled, not run on hardware
//
#include <hip/hip_runtime.h>
#include <hip/hip_bf16.h>

// Problem constants (match reference)
#define DV 512
#define SV 2048
#define HV 8
#define HSV 64
#define VV 32000
#define BBV 4
#define NROWS (BBV * SV) /* 8192 */

typedef __attribute__((ext_vector_type(16))) __bf16 v16bf;
typedef __attribute__((ext_vector_type(16))) unsigned short v16u;
typedef __attribute__((ext_vector_type(8))) float v8f;

static __device__ __forceinline__ unsigned short f2bf(float f) {
  unsigned int u = __float_as_uint(f);
  unsigned int r = (u + 0x7FFFu + ((u >> 16) & 1u)) >> 16;
  return (unsigned short)r;
}

static __device__ __forceinline__ v8f wmma_bf16(v16u a, v16u b, v8f c) {
  return __builtin_amdgcn_wmma_f32_16x16x32_bf16(
      false, __builtin_bit_cast(v16bf, a),
      false, __builtin_bit_cast(v16bf, b),
      (short)0, c, false, false);
}

// ---------------------------------------------------------------------------
// Weight conversion: f32 -> bf16, pre-swizzled into WMMA B-register layout:
//   tile (kt = k/32, nt = n/16), flat tile = kt*Ntiles + nt
//   within tile: lane (0..31) stores 16 contiguous halves; element t of lane
//   holds W[kt*32 + (lane>>4)*16 + t][nt*16 + (lane&15)]
// so in the GEMM each lane loads its whole B operand with one 32B load.
// ---------------------------------------------------------------------------
__global__ __launch_bounds__(256) void convert_qkv_kernel(
    const float* __restrict__ Wq, const float* __restrict__ Wk,
    const float* __restrict__ Wv, unsigned short* __restrict__ wqkv) {
  int tid = blockIdx.x * 256 + threadIdx.x;  // 3*512*512 exact
  int m = tid / (DV * DV);
  int r = tid % (DV * DV);
  int tile = r >> 9;         // 512 tiles per matrix (16 ktiles x 32 ntiles)
  int rr = r & 511;
  int lane = rr >> 4, t = rr & 15;
  int kt = tile >> 5, nt = tile & 31;
  int d = kt * 32 + ((lane >> 4) << 4) + t;  // k index (= input dim d)
  int c = nt * 16 + (lane & 15);             // output col = h*64+e
  int h = c >> 6, e = c & 63;
  const float* W = (m == 0) ? Wq : ((m == 1) ? Wk : Wv);
  wqkv[tid] = f2bf(W[((size_t)h * DV + d) * HSV + e]);
}

__global__ __launch_bounds__(256) void convert_wout_kernel(
    const float* __restrict__ Wout, unsigned short* __restrict__ wout) {
  int tid = blockIdx.x * 256 + threadIdx.x;  // 512*32000 exact
  int tile = tid >> 9;                       // 16 ktiles x 2000 ntiles
  int rr = tid & 511;
  int lane = rr >> 4, t = rr & 15;
  int kt = tile / 2000, nt = tile % 2000;
  int d = kt * 32 + ((lane >> 4) << 4) + t;
  int c = nt * 16 + (lane & 15);
  wout[tid] = f2bf(Wout[(size_t)d * VV + c]);
}

// ---------------------------------------------------------------------------
// Embedding: x[row,d] = tok_emb[index[row],d] + pos_emb[row%S,d]  (bf16)
// ---------------------------------------------------------------------------
__global__ __launch_bounds__(256) void embed_kernel(
    const int* __restrict__ index, const float* __restrict__ tok,
    const float* __restrict__ pos, unsigned short* __restrict__ x) {
  int tid = blockIdx.x * 256 + threadIdx.x;  // 8192*512 exact
  int row = tid >> 9;
  int d = tid & 511;
  int s = row & (SV - 1);
  x[tid] = f2bf(tok[(size_t)index[row] * DV + d] + pos[(size_t)s * DV + d]);
}

// ---------------------------------------------------------------------------
// WMMA GEMM mainloop: C[128x128] += A[128xK] * Bsw[Kx(16*Ntiles)]
// A staged through LDS (64-wide K stages), B read directly from the
// pre-swizzled global layout (one 32B load per lane per tile; L2-resident).
// 256 threads = 8 waves; wave w -> rows (w>>2)*64..+64, cols (w&3)*32..+32.
// ---------------------------------------------------------------------------
#define BM 128
#define BN 128
#define BKK 64

static __device__ __forceinline__ void gemm_mainloop(
    const unsigned short* __restrict__ Ag, const unsigned short* __restrict__ Bsw,
    int lda, int Ntiles, int K, int row0, int col0, v8f acc[4][2]) {
  __shared__ alignas(16) unsigned short As[BM * BKK];
  const int tid = threadIdx.x;
  const int lane = tid & 31;
  const int wave = tid >> 5;
  const int half = lane >> 4;
  const int lr = lane & 15;
  const int wrow = (wave >> 2) * 64;
  const int wcol = (wave & 3) * 32;
  const v16u* Bv = (const v16u*)Bsw;
  const int nt0 = (col0 + wcol) >> 4;
  for (int i = 0; i < 4; i++)
    for (int j = 0; j < 2; j++)
      acc[i][j] = (v8f){0.f, 0.f, 0.f, 0.f, 0.f, 0.f, 0.f, 0.f};

  for (int k0 = 0; k0 < K; k0 += BKK) {
    __syncthreads();
    // Stage A block: 1024 16-byte chunks over 256 threads.
#pragma unroll
    for (int it = 0; it < 4; ++it) {
      int lin = tid + it * 256;
      int ra = lin >> 3, ca = (lin & 7) << 3;
      *(uint4*)&As[ra * BKK + ca] =
          *(const uint4*)&Ag[(size_t)(row0 + ra) * lda + k0 + ca];
    }
    // Prefetch next A stage into cache (global_prefetch_b8).
    if (k0 + BKK < K) {
      int ra = tid >> 1, ca = (tid & 1) << 5;
      __builtin_prefetch(&Ag[(size_t)(row0 + ra) * lda + k0 + BKK + ca], 0, 1);
    }
    __syncthreads();

#pragma unroll
    for (int ch = 0; ch < 2; ++ch) {
      const int kt = (k0 >> 5) + ch;
      v16u a[4];
#pragma unroll
      for (int i = 0; i < 4; i++) {
        const unsigned short* ap = &As[(wrow + i * 16 + lr) * BKK + ch * 32];
        v16u av;
#pragma unroll
        for (int j = 0; j < 8; j++) {
          int kk = (half << 4) + (j << 1);
          unsigned int t = *(const unsigned int*)&ap[kk];
          av[2 * j] = (unsigned short)t;
          av[2 * j + 1] = (unsigned short)(t >> 16);
        }
        a[i] = av;
      }
      v16u b[2];
#pragma unroll
      for (int j = 0; j < 2; j++)
        b[j] = Bv[((size_t)kt * Ntiles + nt0 + j) * 32 + lane];
#pragma unroll
      for (int i = 0; i < 4; i++)
#pragma unroll
        for (int j = 0; j < 2; j++) acc[i][j] = wmma_bf16(a[i], b[j], acc[i][j]);
    }
  }
}

// ---------------------------------------------------------------------------
// QKV projection: Y = x @ Wm, scatter to q[bh,s,e] / k[bh,s,e] / vT[bh,e,s]
// ---------------------------------------------------------------------------
__global__ __launch_bounds__(256) void gemm_qkv_kernel(
    const unsigned short* __restrict__ x, const unsigned short* __restrict__ wqkv,
    unsigned short* __restrict__ qg, unsigned short* __restrict__ kg,
    unsigned short* __restrict__ vTg) {
  const int m = blockIdx.z;
  const int row0 = blockIdx.y * BM;
  const int col0 = blockIdx.x * BN;
  v8f acc[4][2];
  gemm_mainloop(x, wqkv + (size_t)m * DV * DV, DV, DV / 16, DV, row0, col0, acc);

  const int lane = threadIdx.x & 31, wave = threadIdx.x >> 5;
  const int half = lane >> 4, lr = lane & 15;
  const int wrow = (wave >> 2) * 64, wcol = (wave & 3) * 32;
  unsigned short* dst01 = (m == 0) ? qg : kg;
  for (int i = 0; i < 4; i++)
    for (int j = 0; j < 2; j++) {
      int Cc = col0 + wcol + j * 16 + lr;
      int h = Cc >> 6, e2 = Cc & 63;
#pragma unroll
      for (int e = 0; e < 8; e++) {
        int R = row0 + wrow + i * 16 + e + 8 * half;
        int b = R >> 11, s = R & (SV - 1);
        int bh = b * HV + h;
        unsigned short val = f2bf(acc[i][j][e]);
        if (m == 2)
          vTg[((size_t)bh * HSV + e2) * SV + s] = val;
        else
          dst01[((size_t)bh * SV + s) * HSV + e2] = val;
      }
    }
}

// ---------------------------------------------------------------------------
// Flash attention, one (b,h) x 128-row block per workgroup; each wave owns 16
// rows. Streams 32-key blocks with online softmax; WMMA for q·kT and P·V.
// ---------------------------------------------------------------------------
__global__ __launch_bounds__(256) void attention_kernel(
    const unsigned short* __restrict__ qg, const unsigned short* __restrict__ kg,
    const unsigned short* __restrict__ vTg, unsigned short* __restrict__ attn) {
  __shared__ alignas(4) unsigned short Plds[8 * 16 * 32];  // 1KB per wave
  const int bh = blockIdx.y;
  const int b = bh / HV, h = bh % HV;
  const int tid = threadIdx.x;
  const int lane = tid & 31, wave = tid >> 5;
  const int half = lane >> 4, lr = lane & 15;
  const int row0 = blockIdx.x * 128 + wave * 16;  // s-index of wave's rows
  unsigned short* wlds = &Plds[wave * 512];

  // Load q A-operands once (rows row0..row0+15, K = 64 in two 32-chunks)
  const unsigned short* qrow = qg + ((size_t)bh * SV + row0 + lr) * HSV;
  v16u aq[2];
#pragma unroll
  for (int ch = 0; ch < 2; ch++) {
    v16u av;
#pragma unroll
    for (int j = 0; j < 8; j++) {
      int kk = ch * 32 + (half << 4) + (j << 1);
      unsigned int t = *(const unsigned int*)&qrow[kk];
      av[2 * j] = (unsigned short)t;
      av[2 * j + 1] = (unsigned short)(t >> 16);
    }
    aq[ch] = av;
  }

  v8f Oc[4];
  for (int ct = 0; ct < 4; ct++)
    Oc[ct] = (v8f){0.f, 0.f, 0.f, 0.f, 0.f, 0.f, 0.f, 0.f};
  float mrun[8], lrun[8];
#pragma unroll
  for (int e = 0; e < 8; e++) { mrun[e] = -1e30f; lrun[e] = 0.f; }

  const int kb_end = ((row0 + 15) >> 5) + 1;  // causal bound
  for (int kb = 0; kb < kb_end; ++kb) {
    const int key0 = kb * 32;
    // scores S[16x32] as two 16x16 C-tiles, K=64 over two chunks
    v8f s0 = (v8f){0.f, 0.f, 0.f, 0.f, 0.f, 0.f, 0.f, 0.f};
    v8f s1 = s0;
#pragma unroll
    for (int ch = 0; ch < 2; ++ch) {
      const v16u bk0 = *(const v16u*)&kg[((size_t)bh * SV + key0 + lr) * HSV +
                                         ch * 32 + half * 16];
      const v16u bk1 = *(const v16u*)&kg[((size_t)bh * SV + key0 + 16 + lr) * HSV +
                                         ch * 32 + half * 16];
      s0 = wmma_bf16(aq[ch], bk0, s0);
      s1 = wmma_bf16(aq[ch], bk1, s1);
    }
    // scale + causal mask + online softmax (rows live in half-wave lanes)
    float p0[8], p1[8];
#pragma unroll
    for (int e = 0; e < 8; e++) {
      int rowg = row0 + e + 8 * half;
      float v0 = s0[e] * 0.125f;
      if (key0 + lr > rowg) v0 = -1e30f;
      float v1 = s1[e] * 0.125f;
      if (key0 + 16 + lr > rowg) v1 = -1e30f;
      float bm = fmaxf(v0, v1);
      bm = fmaxf(bm, __shfl_xor(bm, 1));
      bm = fmaxf(bm, __shfl_xor(bm, 2));
      bm = fmaxf(bm, __shfl_xor(bm, 4));
      bm = fmaxf(bm, __shfl_xor(bm, 8));
      float mnew = fmaxf(mrun[e], bm);
      float corr = __expf(mrun[e] - mnew);
      float e0 = __expf(v0 - mnew);
      float e1 = __expf(v1 - mnew);
      float rs = e0 + e1;
      rs += __shfl_xor(rs, 1);
      rs += __shfl_xor(rs, 2);
      rs += __shfl_xor(rs, 4);
      rs += __shfl_xor(rs, 8);
      lrun[e] = lrun[e] * corr + rs;
      mrun[e] = mnew;
      for (int ct = 0; ct < 4; ct++) Oc[ct][e] *= corr;
      p0[e] = e0;
      p1[e] = e1;
    }
    // Relayout P (C-layout) -> A-layout through this wave's LDS slice
#pragma unroll
    for (int e = 0; e < 8; e++) {
      int mrow = e + 8 * half;
      wlds[mrow * 32 + lr] = f2bf(p0[e]);
      wlds[mrow * 32 + 16 + lr] = f2bf(p1[e]);
    }
    asm volatile("s_wait_dscnt 0" ::: "memory");  // intra-wave LDS RAW
    v16u pA;
#pragma unroll
    for (int j = 0; j < 8; j++) {
      int kk = (half << 4) + (j << 1);
      unsigned int t = *(const unsigned int*)&wlds[lr * 32 + kk];
      pA[2 * j] = (unsigned short)t;
      pA[2 * j + 1] = (unsigned short)(t >> 16);
    }
    // O += P @ V  (V as B from transposed vT: contiguous 32B per lane)
#pragma unroll
    for (int ct = 0; ct < 4; ct++) {
      const v16u bv = *(const v16u*)&vTg[((size_t)bh * HSV + ct * 16 + lr) * SV +
                                         key0 + half * 16];
      Oc[ct] = wmma_bf16(pA, bv, Oc[ct]);
    }
  }
  // finalize: divide by row sum, store concat-heads bf16 [row, h*64+e]
  for (int ct = 0; ct < 4; ct++) {
#pragma unroll
    for (int e = 0; e < 8; e++) {
      int rowg = row0 + e + 8 * half;
      float val = Oc[ct][e] / lrun[e];
      attn[((size_t)b * SV + rowg) * DV + h * HSV + ct * 16 + lr] = f2bf(val);
    }
  }
}

// ---------------------------------------------------------------------------
// Output projection: logits[8192, 32000] = attn_bf16 @ Wout_sw, f32 out
// ---------------------------------------------------------------------------
__global__ __launch_bounds__(256) void gemm_logits_kernel(
    const unsigned short* __restrict__ attn, const unsigned short* __restrict__ wout,
    float* __restrict__ out) {
  const int row0 = blockIdx.y * BM;
  const int col0 = blockIdx.x * BN;
  v8f acc[4][2];
  gemm_mainloop(attn, wout, DV, VV / 16, DV, row0, col0, acc);

  const int lane = threadIdx.x & 31, wave = threadIdx.x >> 5;
  const int half = lane >> 4, lr = lane & 15;
  const int wrow = (wave >> 2) * 64, wcol = (wave & 3) * 32;
  for (int i = 0; i < 4; i++)
    for (int j = 0; j < 2; j++) {
      int Cc = col0 + wcol + j * 16 + lr;
#pragma unroll
      for (int e = 0; e < 8; e++) {
        int R = row0 + wrow + i * 16 + e + 8 * half;
        out[(size_t)R * VV + Cc] = acc[i][j][e];
      }
    }
}

// ---------------------------------------------------------------------------
// Loss: per-row log-softmax NLL, mean via atomic add
// ---------------------------------------------------------------------------
__global__ void zero_loss_kernel(float* loss) { *loss = 0.f; }

__global__ __launch_bounds__(256) void loss_kernel(
    const float* __restrict__ logits, const int* __restrict__ target,
    float* __restrict__ loss) {
  __shared__ float red[256];
  const int row = blockIdx.x;
  const int tid = threadIdx.x;
  const float* lp = logits + (size_t)row * VV;
  float m = -3.0e38f;
  for (int i = tid; i < VV; i += 256) m = fmaxf(m, lp[i]);
  red[tid] = m;
  __syncthreads();
  for (int s = 128; s > 0; s >>= 1) {
    if (tid < s) red[tid] = fmaxf(red[tid], red[tid + s]);
    __syncthreads();
  }
  float rowmax = red[0];
  __syncthreads();
  float sum = 0.f;
  for (int i = tid; i < VV; i += 256) sum += __expf(lp[i] - rowmax);
  red[tid] = sum;
  __syncthreads();
  for (int s = 128; s > 0; s >>= 1) {
    if (tid < s) red[tid] += red[tid + s];
    __syncthreads();
  }
  if (tid == 0) {
    float logZ = __logf(red[0]) + rowmax;
    int t = target[row];
    atomicAdd(loss, (logZ - lp[t]) * (1.0f / (float)NROWS));
  }
}

// ---------------------------------------------------------------------------
extern "C" void kernel_launch(void* const* d_in, const int* in_sizes, int n_in,
                              void* d_out, int out_size, void* d_ws, size_t ws_size,
                              hipStream_t stream) {
  const int* index = (const int*)d_in[0];
  const int* target = (const int*)d_in[1];
  const float* tok = (const float*)d_in[2];
  const float* pos = (const float*)d_in[3];
  const float* Wq = (const float*)d_in[4];
  const float* Wk = (const float*)d_in[5];
  const float* Wv = (const float*)d_in[6];
  const float* Wout = (const float*)d_in[7];
  float* out = (float*)d_out;
  float* loss = out + (size_t)NROWS * VV;

  unsigned short* ws = (unsigned short*)d_ws;
  unsigned short* x = ws;                             // 8192*512
  unsigned short* wqkv = x + (size_t)NROWS * DV;      // 3*512*512 (swizzled)
  unsigned short* wout = wqkv + (size_t)3 * DV * DV;  // 512*32000 (swizzled)
  unsigned short* qg = wout + (size_t)DV * VV;        // 8192*512
  unsigned short* kg = qg + (size_t)NROWS * DV;       // 8192*512
  unsigned short* vT = kg + (size_t)NROWS * DV;       // 8192*512
  unsigned short* attn = vT + (size_t)NROWS * DV;     // 8192*512

  zero_loss_kernel<<<1, 1, 0, stream>>>(loss);
  convert_qkv_kernel<<<(3 * DV * DV) / 256, 256, 0, stream>>>(Wq, Wk, Wv, wqkv);
  convert_wout_kernel<<<(DV * VV) / 256, 256, 0, stream>>>(Wout, wout);
  embed_kernel<<<(NROWS * DV) / 256, 256, 0, stream>>>(index, tok, pos, x);
  gemm_qkv_kernel<<<dim3(DV / BN, NROWS / BM, 3), 256, 0, stream>>>(x, wqkv, qg, kg, vT);
  attention_kernel<<<dim3(SV / 128, BBV * HV), 256, 0, stream>>>(qg, kg, vT, attn);
  gemm_logits_kernel<<<dim3(VV / BN, NROWS / BM), 256, 0, stream>>>(attn, wout, out);
  loss_kernel<<<NROWS, 256, 0, stream>>>(out, target, loss);
}
